// MultiheadSelfAttention_46866683134418
// MI455X (gfx1250) — matrix operand order repaired and checked by reading, not compiled, not run on hardware
//
#include <hip/hip_runtime.h>

// ---------------------------------------------------------------------------
// MHSA forward for MI455X (gfx1250): f16 WMMA (16x16x32, f32 accum) everywhere,
// GEMM tiles staged by the Tensor Data Mover (TDM) with double buffering.
// ---------------------------------------------------------------------------

typedef __attribute__((ext_vector_type(16))) _Float16 v16h;
typedef __attribute__((ext_vector_type(8)))  _Float16 h8;
typedef __attribute__((ext_vector_type(4)))  _Float16 h4;
typedef __attribute__((ext_vector_type(8)))  float    v8f;
typedef __attribute__((ext_vector_type(4)))  unsigned u32x4;
typedef __attribute__((ext_vector_type(8)))  int      i32x8;
typedef __attribute__((ext_vector_type(4)))  int      i32x4;

union AF   { v16h v; h8 h[2]; };
union U32H { unsigned u; _Float16 f[2]; };

constexpr int DM   = 1024;   // d_model
constexpr int NH   = 16;     // heads
constexpr int HD   = 64;     // head dim
constexpr int B_   = 2;      // batch
constexpr int S_   = 2048;   // seq
constexpr int MTOT = B_ * S_;

__device__ __forceinline__ v8f wmma16(v16h a, v16h b, v8f c) {
  // D = A(16x32 f16) * B(32x16 f16) + C(16x16 f32)
  return __builtin_amdgcn_wmma_f32_16x16x32_f16(false, a, false, b,
                                                (short)0, c, false, false);
}

// ---------------------------------------------------------------------------
// TDM 2D tile load: global f16 tile (tile_d1 rows x tile_d0 elems, row stride
// stride_d0 elems) -> LDS with pad_enable producing +8-half row padding
// (interval = 16 DWORDs per row, amount = 4 DWORDs), i.e. LDS stride 40 halves.
// D# bit layout per CDNA5 ISA 8.3/8.4.
// ---------------------------------------------------------------------------
__device__ __forceinline__ void tdm_load_tile(const _Float16* gptr, _Float16* lptr,
                                              unsigned tile_d0, unsigned tile_d1,
                                              unsigned stride_d0)
{
  unsigned lds_off = (unsigned)(size_t)(void*)lptr;   // flat addr low 32b = LDS offset
  unsigned long long ga = (unsigned long long)(size_t)gptr;

  u32x4 g0;
  g0[0] = 1u;                                          // count=1 (valid user D#)
  g0[1] = lds_off;                                     // lds_addr (bytes)
  g0[2] = (unsigned)(ga & 0xFFFFFFFFu);                // global_addr[31:0]
  g0[3] = (unsigned)((ga >> 32) & 0x01FFFFFFu)         // global_addr[56:32]
        | (2u << 30);                                  // type = 2 ("image")

  const unsigned w0 = (1u << 16)   // data_size = 2 bytes
                    | (1u << 20)   // pad_enable
                    | (3u << 22)   // pad_interval: code 3 = 16 DWORDs (= 32 halves/row)
                    | (3u << 25);  // pad_amount:   code 3 = 4 DWORDs  (= 8 halves)
  i32x8 g1;
  g1[0] = (int)w0;
  g1[1] = (int)((stride_d0 & 0xFFFFu) << 16);                     // tensor_dim0[15:0]
  g1[2] = (int)((stride_d0 >> 16) | ((tile_d1 & 0xFFFFu) << 16)); // dim0 hi | tensor_dim1 lo
  g1[3] = (int)(tile_d0 << 16);                                   // tensor_dim1 hi | tile_dim0
  g1[4] = (int)tile_d1;                                           // tile_dim1 | tile_dim2=0
  g1[5] = (int)stride_d0;                                         // tensor_dim0_stride lo
  g1[6] = 0;                                                      // stride hi | dim1_stride lo
  g1[7] = 0;

  i32x4 z4 = {0, 0, 0, 0};
#if __has_include(<hip/amd_detail/amd_gfx1250_TDM.h>)
  i32x8 z8 = {0, 0, 0, 0, 0, 0, 0, 0};
  __builtin_amdgcn_tensor_load_to_lds(g0, g1, z4, z4, z8, 0);
#else
  __builtin_amdgcn_tensor_load_to_lds(g0, g1, z4, z4, 0);
#endif
}

// ---------------------------------------------------------------------------
// f32 -> f16 bulk convert (4 elements / thread)
// ---------------------------------------------------------------------------
__global__ __launch_bounds__(256)
void f32_to_f16(const float* __restrict__ src, _Float16* __restrict__ dst)
{
  const int i = (blockIdx.x * 256 + threadIdx.x) * 4;
  float4 f = *(const float4*)(src + i);
  h4 hv = { (_Float16)f.x, (_Float16)f.y, (_Float16)f.z, (_Float16)f.w };
  *(h4*)(dst + i) = hv;
}

// ---------------------------------------------------------------------------
// GEMM: C[m,n] = sum_k A[m,k] * W[n,k]   (M=4096, N=K=1024), all-f16 operands.
// Block: 256 thr = 8 waves (4M x 2N), tile 128x128, BK=32.
// Tiles staged by TDM (wave 0 issues), double-buffered for DMA/compute overlap.
// ---------------------------------------------------------------------------
constexpr int BM = 128, BN = 128, BK = 32, LDT = BK + 8;  // 40-half rows (16B aligned)

template<bool OUT_F32>
__global__ __launch_bounds__(256)
void wmma_gemm(const _Float16* __restrict__ A, const _Float16* __restrict__ W,
               void* __restrict__ Craw)
{
  __shared__ __align__(16) _Float16 sA[2][BM * LDT];
  __shared__ __align__(16) _Float16 sB[2][BN * LDT];

  const int tid  = threadIdx.x;
  const int lane = tid & 31, wv = tid >> 5;
  const int lm   = lane & 15, lh = lane >> 4;
  const int wm   = wv >> 1,   wn = wv & 1;
  const int m0   = blockIdx.y * BM, n0 = blockIdx.x * BN;

  v8f acc[2][4] = {};

  // prologue: fill buffer 0
  if (wv == 0) {
    tdm_load_tile(A + (size_t)m0 * DM, sA[0], BK, BM, DM);
    tdm_load_tile(W + (size_t)n0 * DM, sB[0], BK, BN, DM);
    __builtin_amdgcn_s_wait_tensorcnt(0);
  }
  __syncthreads();

  int cur = 0;
  for (int k0 = 0; k0 < DM; k0 += BK) {
    // issue DMA for the next k-step into the other buffer (overlaps compute)
    if (k0 + BK < DM && wv == 0) {
      tdm_load_tile(A + (size_t)m0 * DM + (k0 + BK), sA[cur ^ 1], BK, BM, DM);
      tdm_load_tile(W + (size_t)n0 * DM + (k0 + BK), sB[cur ^ 1], BK, BN, DM);
    }

    // ---- gather fragments from current buffer and multiply ----
    AF a[2], b[4];
    #pragma unroll
    for (int mt = 0; mt < 2; ++mt) {
      const _Float16* ap = &sA[cur][(wm * 32 + mt * 16 + lm) * LDT];
      a[mt].h[0] = *(const h8*)(ap + lh * 8);        // K = lh*8 .. lh*8+7
      a[mt].h[1] = *(const h8*)(ap + 16 + lh * 8);   // K = 16+lh*8 ..
    }
    #pragma unroll
    for (int nt = 0; nt < 4; ++nt) {
      const _Float16* bp = &sB[cur][(wn * 64 + nt * 16 + lm) * LDT + lh * 8];
      b[nt].h[0] = *(const h8*)bp;                   // K = lh*8 .. lh*8+7
      b[nt].h[1] = *(const h8*)(bp + 16);            // K = 16+lh*8 ..
    }
    #pragma unroll
    for (int mt = 0; mt < 2; ++mt)
      #pragma unroll
      for (int nt = 0; nt < 4; ++nt)
        acc[mt][nt] = wmma16(a[mt].v, b[nt].v, acc[mt][nt]);

    if (wv == 0) __builtin_amdgcn_s_wait_tensorcnt(0);
    __syncthreads();            // publish next buffer; retire current
    cur ^= 1;
  }

  // ---- epilogue: C element r -> row r + 8*(lane/16), col lane%16 ----
  #pragma unroll
  for (int mt = 0; mt < 2; ++mt)
    #pragma unroll
    for (int nt = 0; nt < 4; ++nt)
      #pragma unroll
      for (int r = 0; r < 8; ++r) {
        int row = m0 + wm * 32 + mt * 16 + r + lh * 8;
        int col = n0 + wn * 64 + nt * 16 + lm;
        float v = acc[mt][nt][r];
        if constexpr (OUT_F32) ((float*)Craw)[(size_t)row * DM + col] = v;
        else ((_Float16*)Craw)[(size_t)row * DM + col] = (_Float16)v;
      }
}

// ---------------------------------------------------------------------------
// RoPE in-place on f16 (b,s,h,d) buffer; one thread per (even,odd) pair.
// ---------------------------------------------------------------------------
__global__ __launch_bounds__(256)
void rope_kernel(_Float16* __restrict__ X, const int* __restrict__ pos)
{
  const int idx = blockIdx.x * 256 + threadIdx.x;       // = ((b*S+s)*16+h)*32 + i
  const int i   = idx & 31;
  const int s   = (idx >> 9) & (S_ - 1);
  const int b   = idx >> 20;

  unsigned* px = (unsigned*)X;
  U32H u; u.u = px[idx];
  float x1 = (float)u.f[0], x2 = (float)u.f[1];
  float p  = (float)pos[b * S_ + s];
  // inv_freq = 10000^(-i/32) = 2^(-i/32 * log2(10000))
  float freq = exp2f(-(float)i * (13.287712379549449f / 32.0f));
  float sn, cs;
  __sincosf(p * freq, &sn, &cs);
  u.f[0] = (_Float16)(x1 * cs - x2 * sn);
  u.f[1] = (_Float16)(x1 * sn + x2 * cs);
  px[idx] = u.u;
}

// ---------------------------------------------------------------------------
// Causal flash attention, one block per (q-tile of 64, head, batch).
// 128 thr = 4 waves; each wave owns 16 query rows. QK^T and P*V on WMMA.
// ---------------------------------------------------------------------------
__global__ __launch_bounds__(128)
void flash_attn(const _Float16* __restrict__ Q, const _Float16* __restrict__ K,
                const _Float16* __restrict__ V, _Float16* __restrict__ Z)
{
  __shared__ __align__(16) _Float16 Vt[64 * 72];        // V tile transposed [d][key]
  __shared__ __align__(16) _Float16 Pl[4 * 16 * 72];    // per-wave P (16 x 64)

  const int t = blockIdx.x, h = blockIdx.y, b = blockIdx.z;
  const int tid = threadIdx.x;
  const int lane = tid & 31, wv = tid >> 5;
  const int lm = lane & 15, lh = lane >> 4;

  const size_t headoff = (size_t)b * S_ * DM + (size_t)h * HD;
  const _Float16* Qh = Q + headoff;
  const _Float16* Kh = K + headoff;
  const _Float16* Vh = V + headoff;

  // Q A-fragments for this wave's 16 rows, kept in registers for the whole pass
  AF qf[2];
  {
    const _Float16* qp = Qh + (size_t)(t * 64 + wv * 16 + lm) * DM;
    #pragma unroll
    for (int c = 0; c < 2; ++c) {
      qf[c].h[0] = *(const h8*)(qp + c * 32 + lh * 8);
      qf[c].h[1] = *(const h8*)(qp + c * 32 + 16 + lh * 8);
    }
  }

  float mrow[8], lrow[8];
  v8f o[4] = {};
  #pragma unroll
  for (int r = 0; r < 8; ++r) { mrow[r] = -3.0e38f; lrow[r] = 0.f; }

  for (int kt = 0; kt <= t; ++kt) {
    // prefetch next K tile rows into cache while we work on this one
    if (kt < t)
      __builtin_prefetch(Kh + (size_t)((kt + 1) * 64 + lane) * DM, 0, 1);

    // ---- stage V tile transposed into LDS ----
    const unsigned* vsrc = (const unsigned*)(Vh + (size_t)(kt * 64) * DM);
    for (int i = tid; i < 64 * 32; i += 128) {
      int jj = i >> 5, du = i & 31;                    // key row, uint column
      U32H u; u.u = vsrc[(size_t)jj * (DM / 2) + du];
      Vt[(du * 2 + 0) * 72 + jj] = u.f[0];
      Vt[(du * 2 + 1) * 72 + jj] = u.f[1];
    }
    __syncthreads();

    // ---- scores = (Q K^T) * 1/sqrt(64); K B-fragments straight from global ----
    v8f sc[4];
    #pragma unroll
    for (int nt = 0; nt < 4; ++nt) {
      v8f cf = {};
      const _Float16* kp = Kh + (size_t)(kt * 64 + nt * 16 + lm) * DM;
      #pragma unroll
      for (int c = 0; c < 2; ++c) {
        AF bf;
        bf.h[0] = *(const h8*)(kp + c * 32 + lh * 8);
        bf.h[1] = *(const h8*)(kp + c * 32 + 16 + lh * 8);
        cf = wmma16(qf[c].v, bf.v, cf);
      }
      sc[nt] = cf * 0.125f;
    }

    // ---- causal mask on the diagonal tile ----
    if (kt == t) {
      #pragma unroll
      for (int nt = 0; nt < 4; ++nt)
        #pragma unroll
        for (int r = 0; r < 8; ++r)
          if (nt * 16 + lm > wv * 16 + r + lh * 8) sc[nt][r] = -3.0e38f;
    }

    // ---- online softmax (rows replicated across each 16-lane half) ----
    float pv[4][8];
    #pragma unroll
    for (int r = 0; r < 8; ++r) {
      float mx = sc[0][r];
      #pragma unroll
      for (int nt = 1; nt < 4; ++nt) mx = fmaxf(mx, sc[nt][r]);
      #pragma unroll
      for (int m = 1; m < 16; m <<= 1) mx = fmaxf(mx, __shfl_xor(mx, m, 32));
      float mnew  = fmaxf(mrow[r], mx);
      float scale = __expf(mrow[r] - mnew);
      mrow[r] = mnew;
      float rs = 0.f;
      #pragma unroll
      for (int nt = 0; nt < 4; ++nt) {
        float p = __expf(sc[nt][r] - mnew);
        pv[nt][r] = p;
        rs += p;
      }
      #pragma unroll
      for (int m = 1; m < 16; m <<= 1) rs += __shfl_xor(rs, m, 32);
      lrow[r] = lrow[r] * scale + rs;
      #pragma unroll
      for (int nt = 0; nt < 4; ++nt) o[nt][r] = o[nt][r] * scale;
    }

    // ---- P: C-layout regs -> f16 A-layout via wave-private LDS ----
    #pragma unroll
    for (int nt = 0; nt < 4; ++nt)
      #pragma unroll
      for (int r = 0; r < 8; ++r)
        Pl[(wv * 16 + r + lh * 8) * 72 + nt * 16 + lm] = (_Float16)pv[nt][r];
    __syncthreads();

    AF pa[2];
    const _Float16* pp = &Pl[(wv * 16 + lm) * 72];
    #pragma unroll
    for (int c = 0; c < 2; ++c) {
      pa[c].h[0] = *(const h8*)(pp + c * 32 + lh * 8);
      pa[c].h[1] = *(const h8*)(pp + c * 32 + 16 + lh * 8);
    }
    // ---- O += P * V ----
    #pragma unroll
    for (int nt = 0; nt < 4; ++nt)
      #pragma unroll
      for (int c = 0; c < 2; ++c) {
        AF vf;
        const _Float16* vp = &Vt[(nt * 16 + lm) * 72 + c * 32 + lh * 8];
        vf.h[0] = *(const h8*)vp;
        vf.h[1] = *(const h8*)(vp + 16);
        o[nt] = wmma16(pa[c].v, vf.v, o[nt]);
      }
    __syncthreads();
  }

  // ---- epilogue: Z[b,s,h,d] = O / l ----
  #pragma unroll
  for (int nt = 0; nt < 4; ++nt)
    #pragma unroll
    for (int r = 0; r < 8; ++r) {
      int row = t * 64 + wv * 16 + r + lh * 8;
      float val = o[nt][r] / lrow[r];
      Z[((size_t)(b * S_) + row) * DM + h * HD + nt * 16 + lm] = (_Float16)val;
    }
}

// ---------------------------------------------------------------------------
extern "C" void kernel_launch(void* const* d_in, const int* in_sizes, int n_in,
                              void* d_out, int out_size, void* d_ws, size_t ws_size,
                              hipStream_t stream) {
  (void)in_sizes; (void)n_in; (void)out_size; (void)ws_size;
  const float* x  = (const float*)d_in[0];
  const int*  pos = (const int*)d_in[1];
  const float* Wq = (const float*)d_in[2];
  const float* Wk = (const float*)d_in[3];
  const float* Wv = (const float*)d_in[4];
  const float* Wo = (const float*)d_in[5];
  float* out = (float*)d_out;

  const size_t NELT = (size_t)MTOT * DM;          // 4M elements (x, Q, K, V, Z)
  const size_t WELT = (size_t)DM * DM;            // 1M elements (weights)
  _Float16* Qws  = (_Float16*)d_ws;
  _Float16* Kws  = Qws  + NELT;
  _Float16* Vws  = Kws  + NELT;
  _Float16* Zws  = Vws  + NELT;
  _Float16* x16  = Zws  + NELT;
  _Float16* Wq16 = x16  + NELT;
  _Float16* Wk16 = Wq16 + WELT;
  _Float16* Wv16 = Wk16 + WELT;
  _Float16* Wo16 = Wv16 + WELT;                   // total 48 MB of d_ws

  // ---- one-time f32 -> f16 down-convert (halves all GEMM HBM traffic) ----
  f32_to_f16<<<(int)(NELT / 1024), 256, 0, stream>>>(x,  x16);
  f32_to_f16<<<(int)(WELT / 1024), 256, 0, stream>>>(Wq, Wq16);
  f32_to_f16<<<(int)(WELT / 1024), 256, 0, stream>>>(Wk, Wk16);
  f32_to_f16<<<(int)(WELT / 1024), 256, 0, stream>>>(Wv, Wv16);
  f32_to_f16<<<(int)(WELT / 1024), 256, 0, stream>>>(Wo, Wo16);

  dim3 gg(DM / BN, MTOT / BM);                    // (8, 32)
  wmma_gemm<false><<<gg, 256, 0, stream>>>(x16, Wq16, Qws);
  wmma_gemm<false><<<gg, 256, 0, stream>>>(x16, Wk16, Kws);
  wmma_gemm<false><<<gg, 256, 0, stream>>>(x16, Wv16, Vws);

  const int rth = B_ * S_ * NH * (HD / 2);        // 2^21 pairs
  rope_kernel<<<rth / 256, 256, 0, stream>>>(Qws, pos);
  rope_kernel<<<rth / 256, 256, 0, stream>>>(Kws, pos);

  flash_attn<<<dim3(S_ / 64, NH, B_), 128, 0, stream>>>(Qws, Kws, Vws, Zws);

  wmma_gemm<true><<<gg, 256, 0, stream>>>(Zws, Wo16, out);
}
